// MoEMLP_51874615001676
// MI455X (gfx1250) — matrix operand order repaired and checked by reading, not compiled, not run on hardware
//
#include <hip/hip_runtime.h>

#define B_   8
#define L_   8192
#define H_   512
#define E_   8
#define F_   1024
#define KCAP 2048
#define TM   64

typedef __attribute__((ext_vector_type(16))) __bf16 v16bf;
typedef __attribute__((ext_vector_type(8)))  float  v8f;

static __device__ __forceinline__ unsigned f2bfu(float f) {
  union { float f; unsigned u; } a; a.f = f;
  return (a.u + 0x7FFFu + ((a.u >> 16) & 1u)) >> 16;   // RNE
}
static __device__ __forceinline__ __bf16 f2bf(float f) {
  union { unsigned short s; __bf16 h; } b;
  b.s = (unsigned short)f2bfu(f);
  return b.h;
}
// branch-free tanh-form GELU (no EXEC divergence)
static __device__ __forceinline__ float fast_gelu(float v) {
  const float u  = 0.7978845608028654f * v * (1.0f + 0.044715f * v * v);
  const float ex = __expf(2.0f * u);
  const float th = 1.0f - __fdividef(2.0f, ex + 1.0f);
  return 0.5f * v * (1.0f + th);
}

// ---------------- Prep: W1 (E,H,F) f32 -> W1t (E,F,H) bf16 ----------------
__global__ __launch_bounds__(256) void k_prep_w1(const float* __restrict__ W1,
                                                 __bf16* __restrict__ W1t) {
  __shared__ float sT[32][33];
  const int tid = threadIdx.x;
  const int e   = blockIdx.x >> 9;
  const int rem = blockIdx.x & 511;
  const int h0  = (rem >> 5) * 32;
  const int f0  = (rem & 31) * 32;
#pragma unroll
  for (int i = 0; i < 4; ++i) {
    const int idx = tid + i * 256;
    const int r = idx >> 5, c = idx & 31;
    sT[r][c] = W1[((size_t)e * H_ + h0 + r) * F_ + f0 + c];
  }
  __syncthreads();
  unsigned* dst = (unsigned*)W1t;
#pragma unroll
  for (int i = 0; i < 2; ++i) {
    const int idx = tid + i * 256;
    const int c = idx >> 4, j = idx & 15;
    const unsigned u = f2bfu(sT[2 * j][c]) | (f2bfu(sT[2 * j + 1][c]) << 16);
    dst[((size_t)e * F_ + f0 + c) * (H_ / 2) + (h0 >> 1) + j] = u;
  }
}

// ---------------- Prep: W2 (E,F,H) f32 -> W2t (E,H,F) bf16 ----------------
__global__ __launch_bounds__(256) void k_prep_w2(const float* __restrict__ W2,
                                                 __bf16* __restrict__ W2t) {
  __shared__ float sT[32][33];
  const int tid = threadIdx.x;
  const int e   = blockIdx.x >> 9;
  const int rem = blockIdx.x & 511;
  const int f0  = (rem >> 4) * 32;
  const int h0  = (rem & 15) * 32;
#pragma unroll
  for (int i = 0; i < 4; ++i) {
    const int idx = tid + i * 256;
    const int r = idx >> 5, c = idx & 31;
    sT[r][c] = W2[((size_t)e * F_ + f0 + r) * H_ + h0 + c];
  }
  __syncthreads();
  unsigned* dst = (unsigned*)W2t;
#pragma unroll
  for (int i = 0; i < 2; ++i) {
    const int idx = tid + i * 256;
    const int c = idx >> 4, j = idx & 15;
    const unsigned u = f2bfu(sT[2 * j][c]) | (f2bfu(sT[2 * j + 1][c]) << 16);
    dst[((size_t)e * H_ + h0 + c) * (F_ / 2) + (f0 >> 1) + j] = u;
  }
}

// ---------------- Stage 1: gate scores + softmax -> St (B,E,L) ----------------
__global__ __launch_bounds__(256) void k_gate(const float* __restrict__ x,
                                              const float* __restrict__ Wg,
                                              float* __restrict__ St) {
  const int lane = threadIdx.x & 31;
  const int wv   = threadIdx.x >> 5;
  const int t    = blockIdx.x * 8 + wv;
  const int b    = t >> 13;
  const int l    = t & (L_ - 1);
  const float* xr = x + (size_t)t * H_;
  float acc[E_];
#pragma unroll
  for (int e = 0; e < E_; ++e) acc[e] = 0.f;
  for (int i = 0; i < H_ / 32; ++i) {
    const int h = lane + i * 32;
    const float xv = xr[h];
    const float* wr = Wg + h * E_;
#pragma unroll
    for (int e = 0; e < E_; ++e) acc[e] += xv * wr[e];
  }
#pragma unroll
  for (int e = 0; e < E_; ++e) {
#pragma unroll
    for (int off = 16; off > 0; off >>= 1) acc[e] += __shfl_xor(acc[e], off, 32);
  }
  float m = acc[0];
#pragma unroll
  for (int e = 1; e < E_; ++e) m = fmaxf(m, acc[e]);
  float s = 0.f;
#pragma unroll
  for (int e = 0; e < E_; ++e) s += __expf(acc[e] - m);
  if (lane < E_) {
    float me = acc[0];
#pragma unroll
    for (int e = 1; e < E_; ++e) me = (lane == e) ? acc[e] : me;
    St[((size_t)(b * E_ + lane)) * L_ + l] = __expf(me - m) / s;
  }
}

// ---------------- Stage 2: exact top-k (bitwise binary search) ----------------
__global__ __launch_bounds__(256) void k_topk(const float* __restrict__ St,
                                              int* __restrict__ Iw,
                                              float* __restrict__ Gw) {
  const int be  = blockIdx.x;
  const int tid = threadIdx.x;
  __shared__ float sVal[L_];
  __shared__ int sCnt, sGt, sEq;
  const float* row = St + (size_t)be * L_;
  for (int i = tid; i < L_; i += 256) sVal[i] = row[i];
  __syncthreads();
  unsigned lo = 0u, hi = 0x3F800001u;
  for (int it = 0; it < 32; ++it) {
    if (hi - lo <= 1u) break;
    const unsigned mid = lo + ((hi - lo) >> 1);
    if (tid == 0) sCnt = 0;
    __syncthreads();
    int c = 0;
    for (int i = tid; i < L_; i += 256)
      if (__float_as_uint(sVal[i]) >= mid) c++;
    atomicAdd(&sCnt, c);
    __syncthreads();
    const int cnt = sCnt;
    __syncthreads();
    if (cnt >= KCAP) lo = mid; else hi = mid;
  }
  if (tid == 0) { sGt = 0; sEq = 0; }
  __syncthreads();
  int* Ir = Iw + (size_t)be * KCAP;
  float* Gr = Gw + (size_t)be * KCAP;
  for (int i = tid; i < L_; i += 256) {
    const float v = sVal[i];
    if (__float_as_uint(v) > lo) {
      const int p = atomicAdd(&sGt, 1);
      Ir[p] = i; Gr[p] = v;
    }
  }
  __syncthreads();
  const int ngt = sGt;
  for (int i = tid; i < L_; i += 256) {
    const float v = sVal[i];
    if (__float_as_uint(v) == lo) {
      const int p = atomicAdd(&sEq, 1);
      const int slot = ngt + p;
      if (slot < KCAP) { Ir[slot] = i; Gr[slot] = v; }
    }
  }
}

// -------- Stage 3: fused gather + GEMM1 + GELU + GEMM2 + gated scatter --------
__global__ __launch_bounds__(256) void k_moe_mlp(
    const float* __restrict__ x,
    const __bf16* __restrict__ W1t,   // (E,F,H) bf16, K(=H) contiguous
    const float* __restrict__ b1,
    const __bf16* __restrict__ W2t,   // (E,H,F) bf16, K(=F) contiguous
    const float* __restrict__ b2,
    const int* __restrict__ Iw, const float* __restrict__ Gw,
    float* __restrict__ out) {
  __shared__ __align__(16) __bf16 sX[TM * 520];    // 64 x 512 (+8)
  __shared__ __align__(16) __bf16 sH[TM * 1032];   // 64 x 1024 (+8)
  __shared__ __align__(16) __bf16 sWt[8192];       // 256 n x 32 k slab
  __shared__ int   sIdx[TM];
  __shared__ float sG[TM];

  const int tid    = threadIdx.x;
  const int lane   = tid & 31;
  const int wv     = tid >> 5;      // 0..7
  const int mg     = wv & 1;        // 2 M groups (2 tiles of 16 rows each)
  const int ng     = wv >> 1;       // 4 N groups of 64 cols
  const int laneN  = lane & 15;
  const int laneHi = lane >> 4;

  const int chunk = blockIdx.x & 31;   // 32 chunks of 64 tokens
  const int be    = blockIdx.x >> 5;
  const int b     = be >> 3;
  const int e     = be & 7;

  if (tid < TM) {
    sIdx[tid] = Iw[(size_t)be * KCAP + chunk * TM + tid];
    sG[tid]   = Gw[(size_t)be * KCAP + chunk * TM + tid];
  }
  __syncthreads();

  // gather 64 token rows, f32 -> bf16
  for (int i = 0; i < 32; ++i) {
    const int idx4 = tid + i * 256;
    const int row  = idx4 >> 7;
    const int c4   = idx4 & 127;
    const float4 v = ((const float4*)(x + ((size_t)b * L_ + sIdx[row]) * H_))[c4];
    __bf16* d = &sX[row * 520 + c4 * 4];
    d[0] = f2bf(v.x); d[1] = f2bf(v.y); d[2] = f2bf(v.z); d[3] = f2bf(v.w);
  }

  uint4* sWv = (uint4*)sWt;
  const v8f vzero = {};
  v8f acc[2][4];

  // ---- GEMM1: hmid = gelu(Xe @ W1 + b1) -> sH ----
  {
    const uint4* Wv = (const uint4*)W1t;   // idx = (e*F + f)*64 + k/8
    uint4 pf[4];
    auto prefetch = [&](int nc, int ks) {
#pragma unroll
      for (int i = 0; i < 4; ++i) {
        const int c = tid + i * 256;         // 1024 16B chunks: 256 n x 4 kq
        const int n = c >> 2, kq = c & 3;
        pf[i] = Wv[((size_t)e * F_ + nc * 256 + n) * 64 + ks * 4 + kq];
      }
    };
    prefetch(0, 0);
    for (int s = 0; s < 64; ++s) {           // 4 nc x 16 ks
      const int nc = s >> 4, ks = s & 15;
      if (ks == 0) {
#pragma unroll
        for (int mi = 0; mi < 2; ++mi)
#pragma unroll
          for (int nt = 0; nt < 4; ++nt) acc[mi][nt] = vzero;
      }
      __syncthreads();
#pragma unroll
      for (int i = 0; i < 4; ++i) sWv[tid + i * 256] = pf[i];
      __syncthreads();
      if (s + 1 < 64) prefetch((s + 1) >> 4, (s + 1) & 15);
      v16bf av[2], bv[4];
#pragma unroll
      for (int mi = 0; mi < 2; ++mi) {
        const __bf16* ar = &sX[(mg * 32 + mi * 16 + laneN) * 520 + ks * 32 + laneHi * 8];
        ((uint4*)&av[mi])[0] = *(const uint4*)ar;
        ((uint4*)&av[mi])[1] = *(const uint4*)(ar + 16);
      }
#pragma unroll
      for (int nt = 0; nt < 4; ++nt) {
        const __bf16* br = &sWt[(ng * 64 + nt * 16 + laneN) * 32 + laneHi * 8];
        ((uint4*)&bv[nt])[0] = *(const uint4*)br;
        ((uint4*)&bv[nt])[1] = *(const uint4*)(br + 16);
      }
#pragma unroll
      for (int mi = 0; mi < 2; ++mi)
#pragma unroll
        for (int nt = 0; nt < 4; ++nt)
          acc[mi][nt] = __builtin_amdgcn_wmma_f32_16x16x32_bf16(
              false, av[mi], false, bv[nt], (short)0, acc[mi][nt], false, false);
      if (ks == 15) {
#pragma unroll
        for (int mi = 0; mi < 2; ++mi) {
#pragma unroll
          for (int nt = 0; nt < 4; ++nt) {
            const int f = nc * 256 + ng * 64 + nt * 16 + laneN;
            const float bias = b1[e * F_ + f];
#pragma unroll
            for (int r = 0; r < 8; ++r) {
              const int row = mg * 32 + mi * 16 + laneHi * 8 + r;
              sH[row * 1032 + f] = f2bf(fast_gelu(acc[mi][nt][r] + bias));
            }
          }
        }
      }
    }
  }

  // ---- GEMM2: out = G * (hmid @ W2 + b2), scatter-add ----
  {
    const uint4* Wv = (const uint4*)W2t;   // idx = (e*H + h)*128 + k/8
    uint4 pf[4];
    auto prefetch = [&](int hc, int ks) {
#pragma unroll
      for (int i = 0; i < 4; ++i) {
        const int c = tid + i * 256;
        const int n = c >> 2, kq = c & 3;
        pf[i] = Wv[((size_t)e * H_ + hc * 256 + n) * 128 + ks * 4 + kq];
      }
    };
    prefetch(0, 0);
    for (int s = 0; s < 64; ++s) {           // 2 hc x 32 ks
      const int hc = s >> 5, ks = s & 31;
      if (ks == 0) {
#pragma unroll
        for (int mi = 0; mi < 2; ++mi)
#pragma unroll
          for (int nt = 0; nt < 4; ++nt) acc[mi][nt] = vzero;
      }
      __syncthreads();
#pragma unroll
      for (int i = 0; i < 4; ++i) sWv[tid + i * 256] = pf[i];
      __syncthreads();
      if (s + 1 < 64) prefetch((s + 1) >> 5, (s + 1) & 31);
      v16bf av[2], bv[4];
#pragma unroll
      for (int mi = 0; mi < 2; ++mi) {
        const __bf16* ar = &sH[(mg * 32 + mi * 16 + laneN) * 1032 + ks * 32 + laneHi * 8];
        ((uint4*)&av[mi])[0] = *(const uint4*)ar;
        ((uint4*)&av[mi])[1] = *(const uint4*)(ar + 16);
      }
#pragma unroll
      for (int nt = 0; nt < 4; ++nt) {
        const __bf16* br = &sWt[(ng * 64 + nt * 16 + laneN) * 32 + laneHi * 8];
        ((uint4*)&bv[nt])[0] = *(const uint4*)br;
        ((uint4*)&bv[nt])[1] = *(const uint4*)(br + 16);
      }
#pragma unroll
      for (int mi = 0; mi < 2; ++mi)
#pragma unroll
        for (int nt = 0; nt < 4; ++nt)
          acc[mi][nt] = __builtin_amdgcn_wmma_f32_16x16x32_bf16(
              false, av[mi], false, bv[nt], (short)0, acc[mi][nt], false, false);
      if (ks == 31) {
#pragma unroll
        for (int mi = 0; mi < 2; ++mi) {
#pragma unroll
          for (int nt = 0; nt < 4; ++nt) {
            const int h = hc * 256 + ng * 64 + nt * 16 + laneN;
            const float bias = b2[e * H_ + h];
#pragma unroll
            for (int r = 0; r < 8; ++r) {
              const int row = mg * 32 + mi * 16 + laneHi * 8 + r;
              const float v = (acc[mi][nt][r] + bias) * sG[row];
              atomicAdd(&out[((size_t)b * L_ + sIdx[row]) * H_ + h], v);
            }
          }
        }
      }
    }
  }
}

extern "C" void kernel_launch(void* const* d_in, const int* in_sizes, int n_in,
                              void* d_out, int out_size, void* d_ws, size_t ws_size,
                              hipStream_t stream) {
  const float* x  = (const float*)d_in[0];
  const float* Wg = (const float*)d_in[1];
  const float* W1 = (const float*)d_in[2];
  const float* b1 = (const float*)d_in[3];
  const float* W2 = (const float*)d_in[4];
  const float* b2 = (const float*)d_in[5];
  float* out = (float*)d_out;

  char* ws = (char*)d_ws;
  size_t off = 0;
  float* St = (float*)(ws + off); off += (size_t)B_ * E_ * L_ * 4;
  int*   Iw = (int*)(ws + off);   off += (size_t)B_ * E_ * KCAP * 4;
  float* Gw = (float*)(ws + off); off += (size_t)B_ * E_ * KCAP * 4;
  __bf16* W1t = (__bf16*)(ws + off); off += (size_t)E_ * F_ * H_ * 2;
  __bf16* W2t = (__bf16*)(ws + off); off += (size_t)E_ * F_ * H_ * 2;

  hipMemsetAsync(d_out, 0, (size_t)out_size * sizeof(float), stream);
  k_prep_w1<<<E_ * (H_ / 32) * (F_ / 32), 256, 0, stream>>>(W1, W1t);
  k_prep_w2<<<E_ * (F_ / 32) * (H_ / 32), 256, 0, stream>>>(W2, W2t);
  k_gate<<<B_ * L_ / 8, 256, 0, stream>>>(x, Wg, St);
  k_topk<<<B_ * E_, 256, 0, stream>>>(St, Iw, Gw);
  k_moe_mlp<<<B_ * E_ * (KCAP / TM), 256, 0, stream>>>(x, W1t, b1, W2t, b2, Iw, Gw, out);
}
